// RKHS_filter_Model_test_23484881175086
// MI455X (gfx1250) — matrix operand order repaired and checked by reading, not compiled, not run on hardware
//
#include <hip/hip_runtime.h>
#include <math.h>

// ---------------- problem constants ----------------
#define NT   2048      // N tokens
#define DIMF 256       // feature dim
#define NH   4         // heads
#define HD   64        // head dim
#define NL   2         // layers
#define EPSF 1e-5f

// ---------------- WMMA types ----------------
typedef __attribute__((ext_vector_type(16))) __bf16 v16bf;
typedef __attribute__((ext_vector_type(8)))  __bf16 v8bf;
typedef __attribute__((ext_vector_type(8)))  float  v8f;

__device__ __forceinline__ __bf16 f2bf(float f) {
  union { float f; unsigned u; } x; x.f = f;
  unsigned r = x.u + 0x7FFFu + ((x.u >> 16) & 1u);
  unsigned short h = (unsigned short)(r >> 16);
  return __builtin_bit_cast(__bf16, h);
}

// =====================================================================
// LDS-staged strided GEMM, bf16 operands, f32 accumulate via v_wmma.
//   D[m,n] = alpha * sum_k A[m,k]*B[k,n] + beta*Cin[m,n] + bias[n]
// REQUIREMENTS (all call sites satisfy): M % 128 == 0, N % 64 == 0,
// K % 32 == 0.  Block = 256 threads = 8 waves; block tile = 128M x 64N;
// wave w computes rows [w*16, w*16+16) x all 64 N columns = 4 WMMA tiles
// per 32-wide k-step.  B tile (32x64) staged once per block per k-step and
// shared by all 8 waves; A tile (128x32) staged for aligned b128 LDS reads.
// All fragment LDS loads are issued BEFORE the WMMA group so the DS
// round-trip latency overlaps with matrix issue (partial dscnt waits).
// Rows padded to 40 bf16 (80 B) -> 16B-aligned vector reads, no 16-lane
// bank conflicts (20 dwords/row covers 16 distinct banks).
// =====================================================================
#define APAD 40
__global__ __launch_bounds__(256)
void k_gemm_bf16(const float* __restrict__ A, long long sam, long long sak,
                 const float* __restrict__ B, long long sbk, long long sbn,
                 const float* __restrict__ Cin, long long scm, long long scn,
                 const float* __restrict__ bias,
                 float* __restrict__ D, long long sdm, long long sdn,
                 int M, int Nn, int Kk, float alpha, float beta)
{
  __shared__ __bf16 As[128][APAD];
  __shared__ __bf16 Bs[64][APAD];

  const int tid  = threadIdx.x;
  const int lane = tid & 31;
  const int wv   = tid >> 5;
  const int r    = lane & 15;
  const int half = lane >> 4;
  const int mblk = blockIdx.y * 128;
  const int nblk = blockIdx.x * 64;

  // staging maps (uniform, no guards)
  const int am = tid >> 1;          // 0..127
  const int ah = tid & 1;           // 16-wide k half
  const int bn = tid >> 2;          // 0..63
  const int bq = tid & 3;           // 8-wide k quarter

  v8f acc0 = {0.f,0.f,0.f,0.f,0.f,0.f,0.f,0.f};
  v8f acc1 = acc0, acc2 = acc0, acc3 = acc0;

  for (int kb = 0; kb < Kk; kb += 32) {
    // ---- stage A (128 x 32) ----
    {
      const float* Ap = A + (long long)(mblk + am) * sam + (long long)(kb + ah * 16) * sak;
      #pragma unroll
      for (int i = 0; i < 16; ++i)
        As[am][ah * 16 + i] = f2bf(Ap[(long long)i * sak]);
      __builtin_prefetch(Ap + 32LL * sak, 0, 0);   // next k-step (speculative)
    }
    // ---- stage B (32 x 64), stored n-major ----
    {
      const float* Bp = B + (long long)(kb + bq * 8) * sbk + (long long)(nblk + bn) * sbn;
      #pragma unroll
      for (int i = 0; i < 8; ++i)
        Bs[bn][bq * 8 + i] = f2bf(Bp[(long long)i * sbk]);
      __builtin_prefetch(Bp + 32LL * sbk, 0, 0);
    }
    __syncthreads();

    // ---- issue ALL fragment LDS loads first ----
    const int arow = wv * 16 + r;
    v8bf a0  = *(const v8bf*)&As[arow][half * 16];
    v8bf a1  = *(const v8bf*)&As[arow][half * 16 + 8];
    v8bf b00 = *(const v8bf*)&Bs[ 0 + r][half * 16];
    v8bf b01 = *(const v8bf*)&Bs[ 0 + r][half * 16 + 8];
    v8bf b10 = *(const v8bf*)&Bs[16 + r][half * 16];
    v8bf b11 = *(const v8bf*)&Bs[16 + r][half * 16 + 8];
    v8bf b20 = *(const v8bf*)&Bs[32 + r][half * 16];
    v8bf b21 = *(const v8bf*)&Bs[32 + r][half * 16 + 8];
    v8bf b30 = *(const v8bf*)&Bs[48 + r][half * 16];
    v8bf b31 = *(const v8bf*)&Bs[48 + r][half * 16 + 8];

    v16bf af  = __builtin_shufflevector(a0,  a1,  0,1,2,3,4,5,6,7,8,9,10,11,12,13,14,15);
    v16bf bf0 = __builtin_shufflevector(b00, b01, 0,1,2,3,4,5,6,7,8,9,10,11,12,13,14,15);
    v16bf bf1 = __builtin_shufflevector(b10, b11, 0,1,2,3,4,5,6,7,8,9,10,11,12,13,14,15);
    v16bf bf2 = __builtin_shufflevector(b20, b21, 0,1,2,3,4,5,6,7,8,9,10,11,12,13,14,15);
    v16bf bf3 = __builtin_shufflevector(b30, b31, 0,1,2,3,4,5,6,7,8,9,10,11,12,13,14,15);

    // ---- 4 back-to-back WMMAs sharing the A fragment ----
    acc0 = __builtin_amdgcn_wmma_f32_16x16x32_bf16(false, af, false, bf0, (short)0, acc0, false, false);
    acc1 = __builtin_amdgcn_wmma_f32_16x16x32_bf16(false, af, false, bf1, (short)0, acc1, false, false);
    acc2 = __builtin_amdgcn_wmma_f32_16x16x32_bf16(false, af, false, bf2, (short)0, acc2, false, false);
    acc3 = __builtin_amdgcn_wmma_f32_16x16x32_bf16(false, af, false, bf3, (short)0, acc3, false, false);
    __syncthreads();
  }

  // ---- epilogue: D = alpha*acc + beta*Cin + bias (exact tiles, no guards) ----
  const int mrow = mblk + wv * 16 + half * 8;
  #pragma unroll
  for (int t = 0; t < 4; ++t) {
    const v8f* accp = (t == 0) ? &acc0 : (t == 1) ? &acc1 : (t == 2) ? &acc2 : &acc3;
    v8f a = *accp;
    int nc = nblk + t * 16 + r;
    float bv = bias ? bias[nc] : 0.f;
    #pragma unroll
    for (int i = 0; i < 8; ++i) {
      int mm = mrow + i;
      float v = alpha * a[i] + bv;
      if (Cin) v += beta * Cin[(long long)mm * scm + (long long)nc * scn];
      D[(long long)mm * sdm + (long long)nc * sdn] = v;
    }
  }
}

// Naive strided f32 GEMM (numerically sensitive Cholesky / solve updates).
__global__ void k_gemm_f32(const float* __restrict__ A, long long sam, long long sak,
                           const float* __restrict__ B, long long sbk, long long sbn,
                           const float* __restrict__ Cin, long long scm, long long scn,
                           float* __restrict__ D, long long sdm, long long sdn,
                           int M, int Nn, int Kk, float alpha, float beta)
{
  int n = blockIdx.x * 16 + (threadIdx.x & 15);
  int mm = blockIdx.y * 16 + (threadIdx.x >> 4);
  if (mm >= M || n >= Nn) return;
  float s = 0.f;
  for (int k = 0; k < Kk; ++k)
    s += A[(long long)mm*sam + (long long)k*sak] * B[(long long)k*sbk + (long long)n*sbn];
  float v = alpha * s;
  if (Cin) v += beta * Cin[(long long)mm*scm + (long long)n*scn];
  D[(long long)mm*sdm + (long long)n*sdn] = v;
}

// =====================================================================
// Elementwise / reduction kernels
// =====================================================================
__global__ void k_gather(const float* __restrict__ fa, const float* __restrict__ fb,
                         const float* __restrict__ ka, const float* __restrict__ kb,
                         const int* __restrict__ matches,
                         float* __restrict__ fI, float* __restrict__ out_forigin,
                         float* __restrict__ xy, float* __restrict__ xy1)
{
  int idx = blockIdx.x * 256 + threadIdx.x;   // NT*DIMF threads
  int n = idx >> 8, d = idx & 255;
  int ia = matches[2*n], ib = matches[2*n+1];
  float v = fb[(long long)ib*DIMF + d] - fa[(long long)ia*DIMF + d];
  fI[idx] = v;
  out_forigin[idx] = v;
  if (d < 2) { xy[n*2+d] = ka[ia*2+d]; xy1[n*2+d] = kb[ib*2+d]; }
}

__global__ void k_kernelmat(const float* __restrict__ xy, const float* beta_p,
                            const float* lam_p, float* __restrict__ Km, float* __restrict__ Am)
{
  int j = blockIdx.x * 16 + (threadIdx.x & 15);
  int i = blockIdx.y * 16 + (threadIdx.x >> 4);
  float beta = log1pf(expf(beta_p[0]));
  float lam  = log1pf(expf(lam_p[0]));
  float dx = xy[2*i] - xy[2*j], dy = xy[2*i+1] - xy[2*j+1];
  float kk = expf(-beta * (dx*dx + dy*dy));
  long long o = (long long)i*NT + j;
  Km[o] = kk;
  Am[o] = kk + (i == j ? lam : 0.f);
}

__global__ void k_rowsum(const float* __restrict__ Km, float* __restrict__ rs)
{
  __shared__ float red[256];
  int row = blockIdx.x;
  float s = 0.f;
  for (int c = threadIdx.x; c < NT; c += 256) s += Km[(long long)row*NT + c];
  red[threadIdx.x] = s; __syncthreads();
  for (int st = 128; st; st >>= 1) { if (threadIdx.x < st) red[threadIdx.x] += red[threadIdx.x+st]; __syncthreads(); }
  if (threadIdx.x == 0) rs[row] = red[0];
}

__global__ void k_scale_rows(float* __restrict__ fG, const float* __restrict__ rs)
{
  int idx = blockIdx.x * 256 + threadIdx.x;   // NT*DIMF
  int n = idx >> 8;
  fG[idx] /= (rs[n] + 1e-6f);
}

__global__ void k_posenc(const float* __restrict__ xy, const float* __restrict__ Wr,
                         float* __restrict__ enc)   // enc: [2][NT][HD]
{
  int idx = blockIdx.x * 256 + threadIdx.x;   // NT*32
  int n = idx >> 5, t = idx & 31;
  float p = xy[2*n]*Wr[t] + xy[2*n+1]*Wr[32+t];
  float c = cosf(p), s = sinf(p);
  enc[(long long)n*HD + 2*t]     = c;
  enc[(long long)n*HD + 2*t + 1] = c;
  enc[(long long)NT*HD + (long long)n*HD + 2*t]     = s;
  enc[(long long)NT*HD + (long long)n*HD + 2*t + 1] = s;
}

// qkv: [NT][768] interleaved (h*192 + d*3 + {q,k,v}); write rope'd Q,K head-major.
__global__ void k_rope(const float* __restrict__ qkv, const float* __restrict__ enc,
                       float* __restrict__ Q, float* __restrict__ K)
{
  int idx = blockIdx.x * 256 + threadIdx.x;   // NH*NT*HD
  int d = idx & 63, n = (idx >> 6) & (NT-1), h = idx >> 17;
  float c = enc[(long long)n*HD + d];
  float s = enc[(long long)NT*HD + (long long)n*HD + d];
  long long base = (long long)n*768 + h*192;
  int dp = d ^ 1;
  float q  = qkv[base + d*3 + 0];
  float qp = qkv[base + dp*3 + 0];
  float k  = qkv[base + d*3 + 1];
  float kp = qkv[base + dp*3 + 1];
  float rq = (d & 1) ? qp : -qp;
  float rk = (d & 1) ? kp : -kp;
  long long o = (long long)h*NT*HD + (long long)n*HD + d;
  Q[o] = q*c + rq*s;
  K[o] = k*c + rk*s;
}

__global__ void k_softmax(const float* __restrict__ S, float* __restrict__ P, int n, int trans)
{
  __shared__ float red[256];
  int row = blockIdx.x;
  float mx = -3.0e38f;
  for (int c = threadIdx.x; c < n; c += 256) {
    float v = trans ? S[(long long)c*n + row] : S[(long long)row*n + c];
    mx = fmaxf(mx, v);
  }
  red[threadIdx.x] = mx; __syncthreads();
  for (int st = 128; st; st >>= 1) { if (threadIdx.x < st) red[threadIdx.x] = fmaxf(red[threadIdx.x], red[threadIdx.x+st]); __syncthreads(); }
  mx = red[0]; __syncthreads();
  float sum = 0.f;
  for (int c = threadIdx.x; c < n; c += 256) {
    float v = trans ? S[(long long)c*n + row] : S[(long long)row*n + c];
    float e = expf(v - mx);
    P[(long long)row*n + c] = e;
    sum += e;
  }
  red[threadIdx.x] = sum; __syncthreads();
  for (int st = 128; st; st >>= 1) { if (threadIdx.x < st) red[threadIdx.x] += red[threadIdx.x+st]; __syncthreads(); }
  float inv = 1.f / red[0];
  for (int c = threadIdx.x; c < n; c += 256) P[(long long)row*n + c] *= inv;
}

__global__ void k_cat(const float* __restrict__ x, const float* __restrict__ m, float* __restrict__ cat)
{
  int idx = blockIdx.x * 256 + threadIdx.x;   // NT*512
  int n = idx >> 9, c = idx & 511;
  cat[idx] = (c < 256) ? x[(long long)n*256 + c] : m[(long long)n*256 + (c-256)];
}

__global__ void k_ln_gelu(float* __restrict__ Y, const float* __restrict__ g,
                          const float* __restrict__ b, int cols)
{
  __shared__ float red[256];
  long long row = blockIdx.x;
  float* y = Y + row * cols;
  float s = 0.f;
  for (int c = threadIdx.x; c < cols; c += 256) s += y[c];
  red[threadIdx.x] = s; __syncthreads();
  for (int st = 128; st; st >>= 1) { if (threadIdx.x < st) red[threadIdx.x] += red[threadIdx.x+st]; __syncthreads(); }
  float mean = red[0] / cols; __syncthreads();
  float s2 = 0.f;
  for (int c = threadIdx.x; c < cols; c += 256) { float d = y[c] - mean; s2 += d*d; }
  red[threadIdx.x] = s2; __syncthreads();
  for (int st = 128; st; st >>= 1) { if (threadIdx.x < st) red[threadIdx.x] += red[threadIdx.x+st]; __syncthreads(); }
  float inv = rsqrtf(red[0] / cols + EPSF);
  for (int c = threadIdx.x; c < cols; c += 256) {
    float x = (y[c] - mean) * inv * g[c] + b[c];
    y[c] = 0.5f * x * (1.f + erff(x * 0.70710678118654752f));
  }
}

__global__ void k_relu(float* x, int n) { int i = blockIdx.x*256+threadIdx.x; if (i < n) x[i] = fmaxf(x[i], 0.f); }
__global__ void k_copy(float* d, const float* s, int n) { int i = blockIdx.x*256+threadIdx.x; if (i < n) d[i] = s[i]; }

__global__ void k_pred(const float* __restrict__ hb, const float* __restrict__ w2,
                       const float* __restrict__ b2, float* __restrict__ out)
{
  int n = blockIdx.x * 256 + threadIdx.x;
  float s = b2[0];
  for (int t = 0; t < 64; ++t) s += hb[(long long)n*64 + t] * w2[t];
  out[n] = 1.f / (1.f + expf(-s));
}

// =====================================================================
// Blocked Cholesky (64-wide panels) of A (NT x NT, lower), in place.
// =====================================================================
__global__ void k_chol_diag(float* __restrict__ Am, int off)
{
  __shared__ float s[64][65];
  int tid = threadIdx.x;   // 64 threads
  for (int j = 0; j < 64; ++j) s[tid][j] = Am[(long long)(off+tid)*NT + off + j];
  __syncthreads();
  for (int j = 0; j < 64; ++j) {
    if (tid == j) s[j][j] = sqrtf(s[j][j]);
    __syncthreads();
    if (tid > j) s[tid][j] /= s[j][j];
    __syncthreads();
    if (tid > j) for (int t = j+1; t <= tid; ++t) s[tid][t] -= s[tid][j] * s[t][j];
    __syncthreads();
  }
  for (int j = 0; j <= tid; ++j) Am[(long long)(off+tid)*NT + off + j] = s[tid][j];
}

__global__ void k_chol_panel(float* __restrict__ Am, int off, int R)
{
  __shared__ float L[64][65];
  int tid = threadIdx.x;   // 64 threads
  for (int j = 0; j < 64; ++j) L[tid][j] = Am[(long long)(off+tid)*NT + off + j];
  __syncthreads();
  int r = off + 64 + blockIdx.x * 64 + tid;
  if (r >= NT) return;
  float* Ar = Am + (long long)r * NT + off;
  for (int j = 0; j < 64; ++j) {
    float s = Ar[j];
    for (int t = 0; t < j; ++t) s -= Ar[t] * L[j][t];
    Ar[j] = s / L[j][j];
  }
}

// In-block triangular solves, 256 RHS columns (thread per column).
__global__ void k_trsv_lower(const float* __restrict__ Am, float* __restrict__ Cb, int off)
{
  int c = threadIdx.x;
  for (int i = 0; i < 64; ++i) {
    const float* Lr = Am + (long long)(off+i)*NT + off;
    float s = Cb[(long long)(off+i)*DIMF + c];
    for (int t = 0; t < i; ++t) s -= Lr[t] * Cb[(long long)(off+t)*DIMF + c];
    Cb[(long long)(off+i)*DIMF + c] = s / Lr[i];
  }
}
__global__ void k_trsv_upper(const float* __restrict__ Am, float* __restrict__ Cb, int off)
{
  int c = threadIdx.x;
  for (int i = 63; i >= 0; --i) {
    float s = Cb[(long long)(off+i)*DIMF + c];
    for (int t = i+1; t < 64; ++t)
      s -= Am[(long long)(off+t)*NT + off + i] * Cb[(long long)(off+t)*DIMF + c];
    Cb[(long long)(off+i)*DIMF + c] = s / Am[(long long)(off+i)*NT + off + i];
  }
}

// =====================================================================
// Host side
// =====================================================================
static void gemm(hipStream_t st, const float* A, long long sam, long long sak,
                 const float* B, long long sbk, long long sbn,
                 const float* Cin, long long scm, long long scn, float beta,
                 const float* bias, float* D, long long sdm, long long sdn,
                 int M, int N, int K, float alpha)
{
  // requires M%128==0, N%64==0, K%32==0 (true for every call in this model)
  dim3 blk(256), grd(N / 64, M / 128);
  k_gemm_bf16<<<grd, blk, 0, st>>>(A, sam, sak, B, sbk, sbn, Cin, scm, scn, bias,
                                   D, sdm, sdn, M, N, K, alpha, beta);
}
static void gemmf32(hipStream_t st, const float* A, long long sam, long long sak,
                    const float* B, long long sbk, long long sbn,
                    const float* Cin, long long scm, long long scn,
                    float* D, long long sdm, long long sdn,
                    int M, int N, int K, float alpha, float beta)
{
  dim3 blk(256), grd((N + 15) / 16, (M + 15) / 16);
  k_gemm_f32<<<grd, blk, 0, st>>>(A, sam, sak, B, sbk, sbn, Cin, scm, scn,
                                  D, sdm, sdn, M, N, K, alpha, beta);
}

struct Ws {
  float *fI, *fG, *xy, *xy1, *rowsum, *enc0, *enc1, *desc0, *desc1, *Cb;
  float *qkv, *Qr, *Kr, *S, *P, *msg, *msg2, *m0p, *m1p, *cat, *y1, *hb, *Km, *Am;
};

static void ffn_res(hipStream_t st, Ws& W, float* x, const float* msgp,
                    const float* w1, const float* b1, const float* g, const float* bg,
                    const float* w2, const float* b2)
{
  k_cat<<<NT*512/256, 256, 0, st>>>(x, msgp, W.cat);
  gemm(st, W.cat, 512, 1, w1, 512, 1, nullptr, 0, 0, 0.f, b1, W.y1, 512, 1, NT, 512, 512, 1.f);
  k_ln_gelu<<<NT, 256, 0, st>>>(W.y1, g, bg, 512);
  gemm(st, W.y1, 512, 1, w2, 256, 1, x, 256, 1, 1.f, b2, x, 256, 1, NT, 256, 512, 1.f);
}

static void self_block(hipStream_t st, Ws& W, float* x, const float* enc,
                       const float* Wqkv, const float* bqkv, const float* Wout, const float* bout,
                       const float* w1, const float* b1, const float* g, const float* bg,
                       const float* w2, const float* b2)
{
  gemm(st, x, 256, 1, Wqkv, 768, 1, nullptr, 0, 0, 0.f, bqkv, W.qkv, 768, 1, NT, 768, 256, 1.f);
  k_rope<<<NH*NT*HD/256, 256, 0, st>>>(W.qkv, enc, W.Qr, W.Kr);
  for (int h = 0; h < NH; ++h) {
    const float* Qh = W.Qr + (long long)h*NT*HD;
    const float* Kh = W.Kr + (long long)h*NT*HD;
    gemm(st, Qh, HD, 1, Kh, 1, HD, nullptr, 0, 0, 0.f, nullptr, W.S, NT, 1, NT, NT, HD, 0.125f);
    k_softmax<<<NT, 256, 0, st>>>(W.S, W.P, NT, 0);
    gemm(st, W.P, NT, 1, W.qkv + h*192 + 2, 768, 3, nullptr, 0, 0, 0.f, nullptr,
         W.msg + h*64, 256, 1, NT, 64, NT, 1.f);
  }
  gemm(st, W.msg, 256, 1, Wout, 256, 1, nullptr, 0, 0, 0.f, bout, W.msg2, 256, 1, NT, 256, 256, 1.f);
  ffn_res(st, W, x, W.msg2, w1, b1, g, bg, w2, b2);
}

extern "C" void kernel_launch(void* const* d_in, const int* in_sizes, int n_in,
                              void* d_out, int out_size, void* d_ws, size_t ws_size,
                              hipStream_t stream)
{
  const float* features_a = (const float*)d_in[0];
  const float* features_b = (const float*)d_in[1];
  const float* kpts_a     = (const float*)d_in[2];
  const float* kpts_b     = (const float*)d_in[3];
  const int*   matches    = (const int*)  d_in[4];
  const float* lambda_p   = (const float*)d_in[5];
  const float* beta_p     = (const float*)d_in[6];
  const float* posenc_Wr  = (const float*)d_in[7];
  const float* s_qkv_w = (const float*)d_in[8],  * s_qkv_b = (const float*)d_in[9];
  const float* s_out_w = (const float*)d_in[10], * s_out_b = (const float*)d_in[11];
  const float* s_f1_w  = (const float*)d_in[12], * s_f1_b  = (const float*)d_in[13];
  const float* s_ln_g  = (const float*)d_in[14], * s_ln_b  = (const float*)d_in[15];
  const float* s_f2_w  = (const float*)d_in[16], * s_f2_b  = (const float*)d_in[17];
  const float* c_qk_w  = (const float*)d_in[18], * c_qk_b  = (const float*)d_in[19];
  const float* c_v_w   = (const float*)d_in[20], * c_v_b   = (const float*)d_in[21];
  const float* c_out_w = (const float*)d_in[22], * c_out_b = (const float*)d_in[23];
  const float* c_f1_w  = (const float*)d_in[24], * c_f1_b  = (const float*)d_in[25];
  const float* c_ln_g  = (const float*)d_in[26], * c_ln_b  = (const float*)d_in[27];
  const float* c_f2_w  = (const float*)d_in[28], * c_f2_b  = (const float*)d_in[29];
  const float* pred_w1 = (const float*)d_in[30], * pred_b1 = (const float*)d_in[31];
  const float* pred_w2 = (const float*)d_in[32], * pred_b2 = (const float*)d_in[33];

  float* out = (float*)d_out;
  float* out_logits  = out;
  float* out_ffusion = out + NT;
  float* out_forigin = out + NT + (long long)NT*DIMF;
  float* out_fglobal = out + NT + 2LL*NT*DIMF;

  // workspace carve-up
  float* w = (float*)d_ws;
  Ws W;
  auto take = [&](long long n) { float* p = w; w += n; return p; };
  W.fI = take((long long)NT*DIMF);  W.fG = take((long long)NT*DIMF);
  W.xy = take(NT*2);  W.xy1 = take(NT*2);  W.rowsum = take(NT);
  W.enc0 = take(2LL*NT*HD);  W.enc1 = take(2LL*NT*HD);
  W.desc0 = take((long long)NT*DIMF);  W.desc1 = take((long long)NT*DIMF);
  W.Cb = take((long long)NT*DIMF);
  W.qkv = take((long long)NT*768);
  W.Qr = take((long long)NH*NT*HD);  W.Kr = take((long long)NH*NT*HD);
  W.S = take((long long)NT*NT);  W.P = take((long long)NT*NT);
  W.msg = take((long long)NT*DIMF);  W.msg2 = take((long long)NT*DIMF);
  W.m0p = take((long long)NT*DIMF);  W.m1p = take((long long)NT*DIMF);
  W.cat = take((long long)NT*512);   W.y1 = take((long long)NT*512);
  W.hb = take((long long)NT*64);
  W.Km = take((long long)NT*NT);  W.Am = take((long long)NT*NT);

  // 1) gather + pairwise kernel + row sums + positional encodings
  k_gather<<<NT*DIMF/256, 256, 0, stream>>>(features_a, features_b, kpts_a, kpts_b,
                                            matches, W.fI, out_forigin, W.xy, W.xy1);
  { dim3 blk(256), grd(NT/16, NT/16);
    k_kernelmat<<<grd, blk, 0, stream>>>(W.xy, beta_p, lambda_p, W.Km, W.Am); }
  k_rowsum<<<NT, 256, 0, stream>>>(W.Km, W.rowsum);
  k_posenc<<<NT*32/256, 256, 0, stream>>>(W.xy,  posenc_Wr, W.enc0);
  k_posenc<<<NT*32/256, 256, 0, stream>>>(W.xy1, posenc_Wr, W.enc1);

  // 2) Cholesky A = L L^T (blocked, 64)
  for (int b = 0; b < NT/64; ++b) {
    int off = b * 64;
    k_chol_diag<<<1, 64, 0, stream>>>(W.Am, off);
    int R = NT - off - 64;
    if (R > 0) {
      k_chol_panel<<<(R + 63)/64, 64, 0, stream>>>(W.Am, off, R);
      gemmf32(stream,
              W.Am + (long long)(off+64)*NT + off, NT, 1,     // L21 (m,k)
              W.Am + (long long)(off+64)*NT + off, 1, NT,     // L21^T (k,n)
              W.Am + (long long)(off+64)*NT + (off+64), NT, 1,
              W.Am + (long long)(off+64)*NT + (off+64), NT, 1,
              R, R, 64, -1.f, 1.f);
    }
  }
  // 3) solve L y = f_i ; L^T C = y  (256 RHS), in Cb
  k_copy<<<NT*DIMF/256, 256, 0, stream>>>(W.Cb, W.fI, NT*DIMF);
  for (int b = 0; b < NT/64; ++b) {
    int off = b * 64;
    if (b > 0)
      gemmf32(stream, W.Am + (long long)off*NT, NT, 1, W.Cb, DIMF, 1,
              W.Cb + (long long)off*DIMF, DIMF, 1,
              W.Cb + (long long)off*DIMF, DIMF, 1, 64, DIMF, off, -1.f, 1.f);
    k_trsv_lower<<<1, 256, 0, stream>>>(W.Am, W.Cb, off);
  }
  for (int b = NT/64 - 1; b >= 0; --b) {
    int off = b * 64;
    int below = NT - off - 64;
    if (below > 0)
      gemmf32(stream, W.Am + (long long)(off+64)*NT + off, 1, NT,
              W.Cb + (long long)(off+64)*DIMF, DIMF, 1,
              W.Cb + (long long)off*DIMF, DIMF, 1,
              W.Cb + (long long)off*DIMF, DIMF, 1, 64, DIMF, below, -1.f, 1.f);
    k_trsv_upper<<<1, 256, 0, stream>>>(W.Am, W.Cb, off);
  }
  // 4) f_global = (K @ C) / (rowsum + 1e-6)
  gemm(stream, W.Km, NT, 1, W.Cb, DIMF, 1, nullptr, 0, 0, 0.f, nullptr,
       W.fG, DIMF, 1, NT, DIMF, NT, 1.f);
  k_scale_rows<<<NT*DIMF/256, 256, 0, stream>>>(W.fG, W.rowsum);

  k_copy<<<NT*DIMF/256, 256, 0, stream>>>(W.desc0, W.fI, NT*DIMF);
  k_copy<<<NT*DIMF/256, 256, 0, stream>>>(W.desc1, W.fG, NT*DIMF);

  // 5) transformer layers
  for (int l = 0; l < NL; ++l) {
    const float* qkvW = s_qkv_w + (long long)l*256*768;
    const float* qkvB = s_qkv_b + (long long)l*768;
    const float* outW = s_out_w + (long long)l*256*256;
    const float* outB = s_out_b + (long long)l*256;
    const float* f1W  = s_f1_w  + (long long)l*512*512;
    const float* f1B  = s_f1_b  + (long long)l*512;
    const float* lnG  = s_ln_g  + (long long)l*512;
    const float* lnB  = s_ln_b  + (long long)l*512;
    const float* f2W  = s_f2_w  + (long long)l*512*256;
    const float* f2B  = s_f2_b  + (long long)l*256;

    self_block(stream, W, W.desc0, W.enc0, qkvW, qkvB, outW, outB, f1W, f1B, lnG, lnB, f2W, f2B);
    self_block(stream, W, W.desc1, W.enc1, qkvW, qkvB, outW, outB, f1W, f1B, lnG, lnB, f2W, f2B);

    const float* cqkW = c_qk_w + (long long)l*256*256, *cqkB = c_qk_b + (long long)l*256;
    const float* cvW  = c_v_w  + (long long)l*256*256, *cvB  = c_v_b  + (long long)l*256;
    const float* coW  = c_out_w+ (long long)l*256*256, *coB  = c_out_b+ (long long)l*256;
    const float* cf1W = c_f1_w + (long long)l*512*512, *cf1B = c_f1_b + (long long)l*512;
    const float* clnG = c_ln_g + (long long)l*512,     *clnB = c_ln_b + (long long)l*512;
    const float* cf2W = c_f2_w + (long long)l*512*256, *cf2B = c_f2_b + (long long)l*256;

    float* qk0 = W.qkv;                    // reuse qkv buffer: 3 x [NT][256]
    float* qk1 = W.qkv + (long long)NT*256;
    float* v0  = W.qkv + 2LL*NT*256;
    float* v1  = W.Qr;                     // reuse rope buffer
    gemm(stream, W.desc0, 256, 1, cqkW, 256, 1, nullptr,0,0,0.f, cqkB, qk0, 256, 1, NT, 256, 256, 1.f);
    gemm(stream, W.desc1, 256, 1, cqkW, 256, 1, nullptr,0,0,0.f, cqkB, qk1, 256, 1, NT, 256, 256, 1.f);
    gemm(stream, W.desc0, 256, 1, cvW,  256, 1, nullptr,0,0,0.f, cvB,  v0,  256, 1, NT, 256, 256, 1.f);
    gemm(stream, W.desc1, 256, 1, cvW,  256, 1, nullptr,0,0,0.f, cvB,  v1,  256, 1, NT, 256, 256, 1.f);
    for (int h = 0; h < NH; ++h) {
      // sim = qk0_h @ qk1_h^T, alpha = qd^2 = HD^-0.5 = 0.125
      gemm(stream, qk0 + h*64, 256, 1, qk1 + h*64, 1, 256, nullptr,0,0,0.f, nullptr,
           W.S, NT, 1, NT, NT, 64, 0.125f);
      k_softmax<<<NT, 256, 0, stream>>>(W.S, W.P, NT, 0);
      gemm(stream, W.P, NT, 1, v1 + h*64, 256, 1, nullptr,0,0,0.f, nullptr,
           W.msg + h*64, 256, 1, NT, 64, NT, 1.f);
      k_softmax<<<NT, 256, 0, stream>>>(W.S, W.P, NT, 1);
      gemm(stream, W.P, NT, 1, v0 + h*64, 256, 1, nullptr,0,0,0.f, nullptr,
           W.msg2 + h*64, 256, 1, NT, 64, NT, 1.f);
    }
    gemm(stream, W.msg,  256, 1, coW, 256, 1, nullptr,0,0,0.f, coB, W.m0p, 256, 1, NT, 256, 256, 1.f);
    gemm(stream, W.msg2, 256, 1, coW, 256, 1, nullptr,0,0,0.f, coB, W.m1p, 256, 1, NT, 256, 256, 1.f);
    ffn_res(stream, W, W.desc0, W.m0p, cf1W, cf1B, clnG, clnB, cf2W, cf2B);
    ffn_res(stream, W, W.desc1, W.m1p, cf1W, cf1B, clnG, clnB, cf2W, cf2B);
  }

  // 6) prediction head + outputs
  gemm(stream, W.desc0, 256, 1, pred_w1, 64, 1, nullptr,0,0,0.f, pred_b1,
       W.hb, 64, 1, NT, 64, 256, 1.f);
  k_relu<<<NT*64/256, 256, 0, stream>>>(W.hb, NT*64);
  k_pred<<<NT/256, 256, 0, stream>>>(W.hb, pred_w2, pred_b2, out_logits);
  k_copy<<<NT*DIMF/256, 256, 0, stream>>>(out_ffusion, W.desc0, NT*DIMF);
  k_copy<<<NT*DIMF/256, 256, 0, stream>>>(out_fglobal, W.fG, NT*DIMF);
}
